// VQVAE_68436008895093
// MI455X (gfx1250) — matrix operand order repaired and checked by reading, not compiled, not run on hardware
//
#include <hip/hip_runtime.h>
#include <stdint.h>

typedef __attribute__((ext_vector_type(16))) __bf16 v16bf;
typedef __attribute__((ext_vector_type(8)))  __bf16 v8bf;
typedef __attribute__((ext_vector_type(8)))  float  v8f;

#define TB 256

__device__ __forceinline__ float block_reduce_sum(float v) {
    __shared__ float red[TB];
    __syncthreads();
    red[threadIdx.x] = v;
    __syncthreads();
    for (int s = TB / 2; s > 0; s >>= 1) {
        if ((int)threadIdx.x < s) red[threadIdx.x] += red[threadIdx.x + s];
        __syncthreads();
    }
    return red[0];
}

// Async global->LDS copy of 16 bytes (CDNA5 GLOBAL_LOAD_ASYNC_TO_LDS_B128,
// tracked by ASYNCcnt). lds_addr = LDS byte address (low 32 bits of the
// generic pointer), gptr = 64-bit global address in a VGPR pair.
__device__ __forceinline__ void async_copy_b128(uint32_t lds_addr, const void* gptr) {
    asm volatile("global_load_async_to_lds_b128 %0, %1, off"
                 :: "v"(lds_addr), "v"(gptr) : "memory");
}
__device__ __forceinline__ void wait_asynccnt0() {
    asm volatile("s_wait_asynccnt 0" ::: "memory");
}

// ---------------------------------------------------------------------------
// Implicit-GEMM convolution on v_wmma_f32_16x16x32_bf16.
//   M = 8*64*64 = 32768 output positions (exact multiple of the 128 tile),
//   N = O channels (multiple of 64), K = C*KH*KW.
//   Block tile 128(M) x 64(N), K step 32; 8 waves, 32x32 per wave.
//   Double-buffered LDS; B tile staged with async-to-LDS copies of
//   pre-converted bf16 weights; A tile im2col-gathered in 3 phases
//   (addresses -> batched loads -> select/convert) to keep loads in flight.
// flags: bit0 = ReLU on input gather, bit1 = ReLU on output, bit2 = add Res.
// ---------------------------------------------------------------------------
template<int KH, int KW, int STRIDE, int PAD>
__global__ __launch_bounds__(TB)
void conv_bf16_wmma(const float* __restrict__ X, const __bf16* __restrict__ Wb,
                    const float* __restrict__ Bias, const float* __restrict__ Res,
                    float* __restrict__ Y,
                    int C, int H, int W, int O, int flags)
{
    constexpr int khw = KH * KW;
    const int Ktot = C * khw;
    const int m0   = blockIdx.x * 128;
    const int n0   = blockIdx.y * 64;
    const int tid  = threadIdx.x;
    const int lane = tid & 31;
    const int wid  = tid >> 5;
    const int wave_m = wid & 3;
    const int wave_n = wid >> 2;
    const int l15     = lane & 15;
    const int halfsel = (lane >= 16) ? 1 : 0;

    __shared__ alignas(16) __bf16 As[2][128][40];  // [buf][m][k], 80B rows
    __shared__ alignas(16) __bf16 Bs[2][64][40];   // [buf][n][k]

    v8f acc[2][2] = {};

    // A-gather coordinates (HW = 4096, OW = 64 -> shifts/masks only)
    const int arow_ld = tid >> 1;
    const int akb     = (tid & 1) * 16;
    const int mth  = m0 + arow_ld;
    const int ab   = mth >> 12;
    const int ap   = mth & 4095;
    const int aoh  = ap >> 6;
    const int aow  = ap & 63;
    const int xbase = ab * C * H * W;          // per-thread constant
    const int ihb  = aoh * STRIDE - PAD;       // input row base
    const int iwb  = aow * STRIDE - PAD;       // input col base
    // B-tile: thread copies 8 bf16 (16B) of one output-channel row
    const int bn_ld = tid & 63;
    const int bkb   = (tid >> 6) * 8;
    const int gn_ld = n0 + bn_ld;
    const bool in_relu = (flags & 1) != 0;

    auto stage = [&](int buf, int k0) {
        // ---- phase 1: addresses + predicates ----
        int  addrs[16];
        bool oks[16];
        #pragma unroll
        for (int j = 0; j < 16; ++j) {
            int kk  = k0 + akb + j;
            int c   = kk / khw;              // constexpr divisor -> mul/shift
            int rem = kk - c * khw;
            int r   = rem / KW;
            int s   = rem - r * KW;
            int ih  = ihb + r;
            int iw  = iwb + s;
            bool ok = (kk < Ktot) && (ih >= 0) && (ih < H) && (iw >= 0) && (iw < W);
            int addr = xbase + (c * H + ih) * W + iw;
            addrs[j] = ok ? addr : 0;
            oks[j]   = ok;
        }
        // ---- phase 2: batched, unconditional loads ----
        float vals[16];
        #pragma unroll
        for (int j = 0; j < 16; ++j) vals[j] = X[addrs[j]];
        // ---- phase 3: select / ReLU / convert / pack, 2x ds_store_b128 ----
        v8bf p0, p1;
        #pragma unroll
        for (int j = 0; j < 16; ++j) {
            float v = oks[j] ? vals[j] : 0.0f;
            if (in_relu) v = fmaxf(v, 0.0f);
            if (j < 8) p0[j] = (__bf16)v; else p1[j - 8] = (__bf16)v;
        }
        *(v8bf*)&As[buf][arow_ld][akb]     = p0;
        *(v8bf*)&As[buf][arow_ld][akb + 8] = p1;

        // ---- B tile: async 16B global->LDS copies of bf16 weights ----
        if (k0 + bkb + 8 <= Ktot) {
            const __bf16* src = Wb + (size_t)gn_ld * Ktot + k0 + bkb;
            async_copy_b128((uint32_t)(uintptr_t)&Bs[buf][bn_ld][bkb], (const void*)src);
            __builtin_prefetch(src + 32, 0, 1);   // next K slice -> global_prefetch
        } else {                                   // conv0 K-tail (Ktot = 48)
            #pragma unroll
            for (int j = 0; j < 8; ++j) {
                int kk = k0 + bkb + j;
                Bs[buf][bn_ld][bkb + j] = (kk < Ktot) ? Wb[(size_t)gn_ld * Ktot + kk]
                                                      : (__bf16)0.0f;
            }
        }
    };

    stage(0, 0);
    wait_asynccnt0();
    __syncthreads();

    int buf = 0;
    for (int k0 = 0; k0 < Ktot; k0 += 32, buf ^= 1) {
        if (k0 + 32 < Ktot) stage(buf ^ 1, k0 + 32);   // overlap with WMMAs below

        // ---- fragments per CDNA5 16-bit layouts ----
        // A 16x32: lanes 0-15 row=l, K{0..7}+{16..23}; lanes 16-31 K{8..15}+{24..31}
        // B 32x16: lanes 0-15 col=l, K{0..7}+{16..23}; lanes 16-31 K{8..15}+{24..31}
        v16bf afrag[2], bfrag[2];
        #pragma unroll
        for (int ms = 0; ms < 2; ++ms) {
            int arow = wave_m * 32 + ms * 16 + l15;
            int koff = halfsel ? 8 : 0;
            v8bf a0 = *(const v8bf*)(&As[buf][arow][koff]);
            v8bf a1 = *(const v8bf*)(&As[buf][arow][koff + 16]);
            #pragma unroll
            for (int i = 0; i < 8; ++i) { afrag[ms][i] = a0[i]; afrag[ms][i + 8] = a1[i]; }
        }
        #pragma unroll
        for (int ns = 0; ns < 2; ++ns) {
            int bcol = wave_n * 32 + ns * 16 + l15;
            int kb2  = halfsel ? 8 : 0;
            v8bf b0 = *(const v8bf*)(&Bs[buf][bcol][kb2]);
            v8bf b1 = *(const v8bf*)(&Bs[buf][bcol][kb2 + 16]);
            #pragma unroll
            for (int i = 0; i < 8; ++i) { bfrag[ns][i] = b0[i]; bfrag[ns][i + 8] = b1[i]; }
        }
        #pragma unroll
        for (int ms = 0; ms < 2; ++ms)
            #pragma unroll
            for (int ns = 0; ns < 2; ++ns)
                acc[ms][ns] = __builtin_amdgcn_wmma_f32_16x16x32_bf16(
                    false, afrag[ms], false, bfrag[ns], (short)0, acc[ms][ns], false, false);

        wait_asynccnt0();
        __syncthreads();
    }

    // ---- epilogue: exact tiling -> no bounds guards ----
    #pragma unroll
    for (int ms = 0; ms < 2; ++ms) {
        #pragma unroll
        for (int ns = 0; ns < 2; ++ns) {
            int gn = n0 + wave_n * 32 + ns * 16 + l15;
            #pragma unroll
            for (int i = 0; i < 8; ++i) {
                int gm  = m0 + wave_m * 32 + ms * 16 + i + (halfsel ? 8 : 0);
                int b   = gm >> 12;
                int pos = gm & 4095;
                long oidx = (((long)b * O + gn) << 12) + pos;
                float v = acc[ms][ns][i];
                if (Bias) v += Bias[gn];
                if (flags & 4) v += Res[oidx];
                if (flags & 2) v = fmaxf(v, 0.0f);
                Y[oidx] = v;
            }
        }
    }
}

// ---------------------------------------------------------------------------
__global__ __launch_bounds__(TB)
void f32_to_bf16(const float* __restrict__ src, __bf16* __restrict__ dst, int n)
{
    int i = blockIdx.x * TB + threadIdx.x;
    if (i < n) dst[i] = (__bf16)src[i];
}

// BatchNorm (training-mode batch stats); HW fixed at 4096.
__global__ __launch_bounds__(TB)
void bn_stats(const float* __restrict__ X, float* __restrict__ mean,
              float* __restrict__ istd, int C)
{
    int c = blockIdx.x;
    float s1 = 0.0f, s2 = 0.0f;
    const int n = 8 * 4096;
    for (int i = threadIdx.x; i < n; i += TB) {
        int b = i >> 12;
        int p = i & 4095;
        float v = X[((b * C + c) << 12) + p];
        s1 += v;
        s2 += v * v;
    }
    s1 = block_reduce_sum(s1);
    s2 = block_reduce_sum(s2);
    if (threadIdx.x == 0) {
        float m   = s1 / (float)n;
        float var = s2 / (float)n - m * m;
        mean[c] = m;
        istd[c] = rsqrtf(var + 1e-5f);
    }
}

__global__ __launch_bounds__(TB)
void bn_apply_relu(float* __restrict__ X, const float* __restrict__ mean,
                   const float* __restrict__ istd, const float* __restrict__ g,
                   const float* __restrict__ beta, int Cmask, int total)
{
    int i = blockIdx.x * TB + threadIdx.x;
    if (i >= total) return;
    int c = (i >> 12) & Cmask;   // C is a power of two
    float v = (X[i] - mean[c]) * istd[c] * g[c] + beta[c];
    X[i] = fmaxf(v, 0.0f);
}

__global__ __launch_bounds__(TB)
void relu_inplace(float* __restrict__ X, int n)
{
    int i = blockIdx.x * TB + threadIdx.x;
    if (i < n) X[i] = fmaxf(X[i], 0.0f);
}

// ---------------------------------------------------------------------------
// Vector quantizer
// ---------------------------------------------------------------------------
__global__ __launch_bounds__(TB)
void code_norms(const float* __restrict__ cb, float* __restrict__ norms)
{
    int k = blockIdx.x * TB + threadIdx.x;
    if (k >= 1024) return;
    float s = 0.0f;
    for (int d = 0; d < 256; ++d) { float v = cb[(k << 8) + d]; s += v * v; }
    norms[k] = s;
}

__global__ __launch_bounds__(TB)
void vq_argmin(const float* __restrict__ S, const float* __restrict__ norms,
               int* __restrict__ idx)
{
    int n = blockIdx.x * TB + threadIdx.x;
    if (n >= 8 * 4096) return;
    int b = n >> 12;
    int p = n & 4095;
    const float* sp = S + (((long)b) << 22) + p;   // b*1024*4096
    float best = 3.0e38f;
    int bi = 0;
    for (int k = 0; k < 1024; ++k) {
        float d = norms[k] - 2.0f * sp[(long)k << 12];
        if (d < best) { best = d; bi = k; }
    }
    idx[n] = bi;
}

__global__ __launch_bounds__(TB)
void vq_gather(const float* __restrict__ Z, const float* __restrict__ cb,
               const int* __restrict__ idx, float* __restrict__ ZQ,
               float* __restrict__ vq_acc)
{
    int n = blockIdx.x;        // over 8*4096 positions
    int d = threadIdx.x;       // D == 256 == TB
    int b = n >> 12;
    int p = n & 4095;
    float q = cb[(idx[n] << 8) + d];
    long zi = (((long)(b << 8) + d) << 12) + p;    // ((b*256+d)*4096)+p
    float z = Z[zi];
    ZQ[zi] = q;                // straight-through forward value == q
    float diff = q - z;
    float s = block_reduce_sum(diff * diff);
    if (threadIdx.x == 0) atomicAdd(vq_acc, s);
}

// ---------------------------------------------------------------------------
// ConvTranspose2d(k=4,s=2,p=1) 256->3 at 64x64 -> 128x128, fused recon MSE.
// ---------------------------------------------------------------------------
__global__ __launch_bounds__(TB)
void deconv_k4s2_mse(const float* __restrict__ Hin, const float* __restrict__ Wt,
                     const float* __restrict__ Bias, const float* __restrict__ Xref,
                     float* __restrict__ Xr, float* __restrict__ recon_acc)
{
    constexpr int C = 256, Hh = 64, Ww = 64, O = 3, OH = 128, OW = 128;
    int i = blockIdx.x * TB + threadIdx.x;
    const int total = 8 * O * OH * OW;
    float ds = 0.0f;
    if (i < total) {
        int ow = i & (OW - 1);
        int t  = i >> 7;
        int oh = t & (OH - 1);
        t >>= 7;
        int o = t % O;
        int b = t / O;
        float sum = Bias[o];
        for (int c = 0; c < C; ++c) {
            const float* hc = Hin + ((long)(b * C + c) * Hh) * Ww;
            const float* wc = Wt + ((long)(c * O + o)) * 16;  // [C][O][4][4]
            #pragma unroll
            for (int ky = 0; ky < 4; ++ky) {
                int iy2 = oh + 1 - ky;
                if (iy2 < 0 || (iy2 & 1)) continue;
                int iy = iy2 >> 1;
                if (iy >= Hh) continue;
                #pragma unroll
                for (int kx = 0; kx < 4; ++kx) {
                    int ix2 = ow + 1 - kx;
                    if (ix2 < 0 || (ix2 & 1)) continue;
                    int ix = ix2 >> 1;
                    if (ix >= Ww) continue;
                    sum += hc[iy * Ww + ix] * wc[ky * 4 + kx];
                }
            }
        }
        Xr[i] = sum;
        float d = sum - Xref[i];
        ds = d * d;
    }
    float s = block_reduce_sum(ds);
    if (threadIdx.x == 0) atomicAdd(recon_acc, s);
}

__global__ void init_acc(float* acc)
{
    if (threadIdx.x < 2) acc[threadIdx.x] = 0.0f;
}

__global__ void finalize_losses(const float* __restrict__ acc, float* __restrict__ out,
                                float inv_nr, float inv_nv)
{
    if (threadIdx.x == 0) {
        float recon = acc[0] * inv_nr;
        float vq    = acc[1] * inv_nv;     // q_loss + 0.25*e_loss = 1.25*mse
        out[0] = recon + 1.25f * vq;
        out[1] = recon;
    }
}

// ---------------------------------------------------------------------------
template<int KH, int KW, int ST, int PD>
static void launch_conv(hipStream_t stream, const float* X, const __bf16* Wb,
                        const float* Bias, const float* Res, float* Y,
                        int C, int H, int W, int O, int flags)
{
    const int M = 8 * 64 * 64;
    dim3 grid(M / 128, O / 64);
    conv_bf16_wmma<KH, KW, ST, PD><<<grid, TB, 0, stream>>>(X, Wb, Bias, Res, Y,
                                                            C, H, W, O, flags);
}

extern "C" void kernel_launch(void* const* d_in, const int* in_sizes, int n_in,
                              void* d_out, int out_size, void* d_ws, size_t ws_size,
                              hipStream_t stream)
{
    (void)in_sizes; (void)n_in; (void)out_size; (void)ws_size;

    const float* x     = (const float*)d_in[0];
    const float* e0_w  = (const float*)d_in[1];
    const float* e0_b  = (const float*)d_in[2];
    const float* bn0_g = (const float*)d_in[3];
    const float* bn0_b = (const float*)d_in[4];
    const float* e1_w  = (const float*)d_in[5];
    const float* e1_b  = (const float*)d_in[6];
    const float* bn1_g = (const float*)d_in[7];
    const float* bn1_b = (const float*)d_in[8];
    const float* e2_w  = (const float*)d_in[9];
    const float* e2_b  = (const float*)d_in[10];
    const float* erw1  = (const float*)d_in[11];
    const float* erb1  = (const float*)d_in[12];
    const float* erw2  = (const float*)d_in[13];
    const float* erb2  = (const float*)d_in[14];
    const float* e3_w  = (const float*)d_in[15];
    const float* e3_b  = (const float*)d_in[16];
    const float* cb    = (const float*)d_in[17];
    const float* d0_w  = (const float*)d_in[18];
    const float* d0_b  = (const float*)d_in[19];
    const float* drw1  = (const float*)d_in[20];
    const float* drb1  = (const float*)d_in[21];
    const float* drw2  = (const float*)d_in[22];
    const float* drb2  = (const float*)d_in[23];
    const float* d1_w  = (const float*)d_in[24];
    const float* d1_b  = (const float*)d_in[25];
    const float* dt_w  = (const float*)d_in[26];
    const float* dt_b  = (const float*)d_in[27];

    const int HW = 4096;
    char* base = (char*)d_ws;
    size_t off = 0;
    auto allocf = [&](size_t n) { float* p = (float*)(base + off); off += n * 4; return p; };
    auto allocb = [&](size_t n) { __bf16* p = (__bf16*)(base + off);
                                  off += ((n * 2 + 15) & ~(size_t)15); return p; };

    float* t0    = allocf((size_t)8 * 256 * HW);
    float* t1    = allocf((size_t)8 * 512 * HW);
    float* t2    = allocf((size_t)8 * 512 * HW);
    float* z     = allocf((size_t)8 * 256 * HW);
    float* tmp   = allocf((size_t)8 * 128 * HW);
    float* zq    = allocf((size_t)8 * 256 * HW);
    float* S     = allocf((size_t)8 * 1024 * HW);
    float* norms = allocf(1024);
    float* mean  = allocf(512);
    float* istd  = allocf(512);
    float* accs  = allocf(2);
    int*   idx   = (int*)allocf(8 * HW);

    // bf16 weight arena
    __bf16* wb_e0   = allocb((size_t)256 * 48);
    __bf16* wb_e1   = allocb((size_t)512 * 2304);
    __bf16* wb_e2   = allocb((size_t)512 * 4608);
    __bf16* wb_erw1 = allocb((size_t)16 * 128 * 4608);
    __bf16* wb_erw2 = allocb((size_t)16 * 512 * 128);
    __bf16* wb_e3   = allocb((size_t)256 * 4608);
    __bf16* wb_cb   = allocb((size_t)1024 * 256);
    __bf16* wb_d0   = allocb((size_t)512 * 2304);
    __bf16* wb_drw1 = allocb((size_t)16 * 128 * 4608);
    __bf16* wb_drw2 = allocb((size_t)16 * 512 * 128);
    __bf16* wb_d1   = allocb((size_t)256 * 4608);

    auto cvt = [&](const float* s, __bf16* d, size_t n) {
        f32_to_bf16<<<((int)n + TB - 1) / TB, TB, 0, stream>>>(s, d, (int)n);
    };
    cvt(e0_w, wb_e0, (size_t)256 * 48);
    cvt(e1_w, wb_e1, (size_t)512 * 2304);
    cvt(e2_w, wb_e2, (size_t)512 * 4608);
    cvt(erw1, wb_erw1, (size_t)16 * 128 * 4608);
    cvt(erw2, wb_erw2, (size_t)16 * 512 * 128);
    cvt(e3_w, wb_e3, (size_t)256 * 4608);
    cvt(cb,   wb_cb, (size_t)1024 * 256);
    cvt(d0_w, wb_d0, (size_t)512 * 2304);
    cvt(drw1, wb_drw1, (size_t)16 * 128 * 4608);
    cvt(drw2, wb_drw2, (size_t)16 * 512 * 128);
    cvt(d1_w, wb_d1, (size_t)256 * 4608);

    init_acc<<<1, 64, 0, stream>>>(accs);

    // ---------------- encoder ----------------
    launch_conv<4, 4, 2, 1>(stream, x, wb_e0, e0_b, nullptr, t0, 3, 128, 128, 256, 0);
    bn_stats<<<256, TB, 0, stream>>>(t0, mean, istd, 256);
    bn_apply_relu<<<(8 * 256 * HW) / TB, TB, 0, stream>>>(
        t0, mean, istd, bn0_g, bn0_b, 255, 8 * 256 * HW);

    launch_conv<3, 3, 1, 1>(stream, t0, wb_e1, e1_b, nullptr, t1, 256, 64, 64, 512, 0);
    bn_stats<<<512, TB, 0, stream>>>(t1, mean, istd, 512);
    bn_apply_relu<<<(8 * 512 * HW) / TB, TB, 0, stream>>>(
        t1, mean, istd, bn1_g, bn1_b, 511, 8 * 512 * HW);

    launch_conv<3, 3, 1, 1>(stream, t1, wb_e2, e2_b, nullptr, t2, 512, 64, 64, 512, 0);

    for (int s = 0; s < 4; ++s) {
        for (int i = 0; i < 4; ++i) {
            int li = s * 4 + i;
            launch_conv<3, 3, 1, 1>(stream, t2, wb_erw1 + (size_t)li * 128 * 4608,
                                    erb1 + li * 128, nullptr, tmp, 512, 64, 64, 128, 1 | 2);
            launch_conv<1, 1, 1, 0>(stream, tmp, wb_erw2 + (size_t)li * 512 * 128,
                                    erb2 + li * 512, t2, t2, 128, 64, 64, 512, 4);
        }
        relu_inplace<<<(8 * 512 * HW) / TB, TB, 0, stream>>>(t2, 8 * 512 * HW);
    }

    launch_conv<3, 3, 1, 1>(stream, t2, wb_e3, e3_b, nullptr, z, 512, 64, 64, 256, 0);

    // ---------------- vector quantizer ----------------
    code_norms<<<4, TB, 0, stream>>>(cb, norms);
    launch_conv<1, 1, 1, 0>(stream, z, wb_cb, nullptr, nullptr, S, 256, 64, 64, 1024, 0);
    vq_argmin<<<(8 * HW) / TB, TB, 0, stream>>>(S, norms, idx);
    vq_gather<<<8 * HW, TB, 0, stream>>>(z, cb, idx, zq, accs + 1);

    // ---------------- decoder ----------------
    launch_conv<3, 3, 1, 1>(stream, zq, wb_d0, d0_b, nullptr, t1, 256, 64, 64, 512, 0);

    for (int s = 0; s < 4; ++s) {
        for (int i = 0; i < 4; ++i) {
            int li = s * 4 + i;
            launch_conv<3, 3, 1, 1>(stream, t1, wb_drw1 + (size_t)li * 128 * 4608,
                                    drb1 + li * 128, nullptr, tmp, 512, 64, 64, 128, 1 | 2);
            launch_conv<1, 1, 1, 0>(stream, tmp, wb_drw2 + (size_t)li * 512 * 128,
                                    drb2 + li * 512, t1, t1, 128, 64, 64, 512, 4);
        }
        relu_inplace<<<(8 * 512 * HW) / TB, TB, 0, stream>>>(t1, 8 * 512 * HW);
    }

    launch_conv<3, 3, 1, 1>(stream, t1, wb_d1, d1_b, nullptr, t0, 512, 64, 64, 256, 2);

    float* xr = (float*)d_out;
    deconv_k4s2_mse<<<(8 * 3 * 128 * 128) / TB, TB, 0, stream>>>(
        t0, dt_w, dt_b, x, xr, accs);

    finalize_losses<<<1, 32, 0, stream>>>(accs, xr + 8 * 3 * 128 * 128,
                                          1.0f / (8.0f * 3 * 128 * 128),
                                          1.0f / (8.0f * 4096 * 256));
}
